// KerasCrossAttentionModule_89962384982254
// MI455X (gfx1250) — matrix operand order repaired and checked by reading, not compiled, not run on hardware
//
#include <hip/hip_runtime.h>

// Problem constants (from reference)
#define DIM_N   4096
#define DIM_C   256
#define BATCH   8

// Tiling
#define TQ       64     // query rows per workgroup
#define TK       32     // kv rows per tile (== WMMA K for the P*V gemm)
#define NWAVE    4      // 1 wave per 16 query rows
#define NTHREADS 128
#define NT       (DIM_N / TK)   // 128 kv tiles

// LDS strides (f16 elements), padded against bank conflicts
#define QSTR 264
#define KSTR 264
#define VSTR 40
#define PSTR 40
#define OSTR 260        // f32 elements (epilogue overlay)

typedef __attribute__((ext_vector_type(16))) _Float16 v16h;
typedef __attribute__((ext_vector_type(8)))  _Float16 v8h;
typedef __attribute__((ext_vector_type(4)))  _Float16 v4h;
typedef __attribute__((ext_vector_type(2)))  _Float16 h2;
typedef __attribute__((ext_vector_type(8)))  float    v8f;
typedef __attribute__((ext_vector_type(4)))  int      i32x4;

// ---------------- async global->LDS copy (gfx1250 path, guarded) ----------------
#if __has_builtin(__builtin_amdgcn_global_load_async_to_lds_b128) && \
    __has_builtin(__builtin_amdgcn_s_wait_asynccnt)
#define USE_ASYNC 1
#else
#define USE_ASYNC 0
#endif

__device__ __forceinline__ void cp16(const _Float16* g, _Float16* l) {
#if USE_ASYNC
    // param0: addrspace(1) int4*  (global source; clang prints AS1 as __device__)
    // param1: addrspace(3) int4*  (LDS destination)
    __builtin_amdgcn_global_load_async_to_lds_b128(
        (__attribute__((address_space(1))) i32x4*)(void*)g,
        (__attribute__((address_space(3))) i32x4*)(void*)l,
        0, 0);
#else
    *(v8h*)l = *(const v8h*)g;
#endif
}
__device__ __forceinline__ void wait_async_16() {
#if USE_ASYNC
    __builtin_amdgcn_s_wait_asynccnt(16);
#endif
}
__device__ __forceinline__ void wait_async_0() {
#if USE_ASYNC
    __builtin_amdgcn_s_wait_asynccnt(0);
#endif
}

// ---------------- WMMA helpers ----------------
__device__ __forceinline__ v8f wmma_f16(v16h a, v16h b, v8f c) {
    return __builtin_amdgcn_wmma_f32_16x16x32_f16(
        false, a, false, b, (short)0, c, false, false);
}

// A-matrix fragment (16x32 f16, ISA 7.12.2): lane holds row M = lane&15.
// VGPR v in 0..7: K = 2*hi8 + 2v,2v+1 (contiguous, matches B fragment).
__device__ __forceinline__ v16h load_afrag(const _Float16* __restrict__ rowp, int hi8) {
    v16h a;
#pragma unroll
    for (int v = 0; v < 8; ++v) {
        int col = 2 * hi8 + 2 * v;
        h2 p = *(const h2*)(rowp + col);
        a[2 * v]     = p.x;
        a[2 * v + 1] = p.y;
    }
    return a;
}

// B-matrix fragment (32x16 f16): lane holds column N = lane&15, K = kb16 + i.
__device__ __forceinline__ v16h load_bfrag(const _Float16* __restrict__ p) {
    v8h lo = *(const v8h*)(p);
    v8h hi = *(const v8h*)(p + 8);
    return __builtin_shufflevector(lo, hi, 0, 1, 2, 3, 4, 5, 6, 7, 8, 9, 10, 11, 12, 13, 14, 15);
}

// =======================================================================
// Pre-pass kernels: materialize f16 Q(+pos)[B,N,C], K(+pos)[B,N,C], V[B,C,N]
// =======================================================================
__global__ __launch_bounds__(256)
void prep_qk(const float* __restrict__ src, const float* __restrict__ pos,
             _Float16* __restrict__ dst) {
    __shared__ float t[64][65];
    const int n0 = blockIdx.x * 64, c0 = blockIdx.y * 64, b = blockIdx.z;
    const int tx = threadIdx.x & 63, ty = threadIdx.x >> 6;   // ty in 0..3
#pragma unroll
    for (int i = 0; i < 16; ++i) {
        int c = ty + i * 4;
        t[c][tx] = src[(b * DIM_C + c0 + c) * DIM_N + n0 + tx];   // coalesced along n
    }
    __syncthreads();
#pragma unroll
    for (int i = 0; i < 16; ++i) {
        int n = ty + i * 4;
        float val = t[tx][n] + pos[(n0 + n) * DIM_C + c0 + tx];
        dst[(b * DIM_N + n0 + n) * DIM_C + c0 + tx] = (_Float16)val; // coalesced along c
    }
}

__global__ __launch_bounds__(256)
void prep_v(const float* __restrict__ src, _Float16* __restrict__ dst) {
    int i = (blockIdx.x * 256 + threadIdx.x) * 4;    // total elems divisible by 1024
    float4 f = *(const float4*)(src + i);
    v4h h = {(_Float16)f.x, (_Float16)f.y, (_Float16)f.z, (_Float16)f.w};
    *(v4h*)(dst + i) = h;
}

// =======================================================================
// Fast path: flash attention on pre-converted f16, async double-buffered
// =======================================================================
#define F_OFF_Q  0
#define F_OFF_K0 (TQ * QSTR * 2)                       // 33792
#define F_OFF_K1 (F_OFF_K0 + TK * KSTR * 2)            // 50688
#define F_OFF_V0 (F_OFF_K1 + TK * KSTR * 2)            // 67584
#define F_OFF_V1 (F_OFF_V0 + DIM_C * VSTR * 2)         // 88064
#define F_OFF_P  (F_OFF_V1 + DIM_C * VSTR * 2)         // 108544
#define F_SMEM   (F_OFF_P + NWAVE * 16 * PSTR * 2)     // 113664; sO overlay 66560 fits

__device__ __forceinline__ void stage_kv(const _Float16* __restrict__ Kf,
                                         const _Float16* __restrict__ Vf,
                                         _Float16* sKb, _Float16* sVb,
                                         int b, int k0, int tid) {
#pragma unroll
    for (int i = 0; i < 8; ++i) {                 // K tile: 32 rows x 512B
        int idx = tid + i * NTHREADS;
        int r = idx >> 5, cc = (idx & 31) << 3;
        cp16(Kf + (b * DIM_N + k0 + r) * DIM_C + cc, sKb + r * KSTR + cc);
    }
#pragma unroll
    for (int i = 0; i < 8; ++i) {                 // V tile: 256 rows x 64B (already [c][n])
        int idx = tid + i * NTHREADS;
        int c = idx >> 2, rc = (idx & 3) << 3;
        cp16(Vf + (b * DIM_C + c) * DIM_N + k0 + rc, sVb + c * VSTR + rc);
    }
}

__global__ __launch_bounds__(NTHREADS)
void xattn_flash_f16(const _Float16* __restrict__ Qf, const _Float16* __restrict__ Kf,
                     const _Float16* __restrict__ Vf, float* __restrict__ gout) {
    __shared__ __align__(16) char smem[F_SMEM];
    _Float16* sQ  = (_Float16*)(smem + F_OFF_Q);
    _Float16* sK0 = (_Float16*)(smem + F_OFF_K0);
    _Float16* sK1 = (_Float16*)(smem + F_OFF_K1);
    _Float16* sV0 = (_Float16*)(smem + F_OFF_V0);
    _Float16* sV1 = (_Float16*)(smem + F_OFF_V1);
    _Float16* sP  = (_Float16*)(smem + F_OFF_P);
    float*    sO  = (float*)smem;

    const int tid    = threadIdx.x;
    const int wave   = tid >> 5;
    const int lane   = tid & 31;
    const int lane15 = lane & 15;
    const int hi8    = (lane >> 4) << 3;
    const int kb16   = (lane >> 4) << 4;

    const int b  = blockIdx.x / (DIM_N / TQ);
    const int q0 = (blockIdx.x % (DIM_N / TQ)) * TQ;

    // ---- prefetch: Q tile + kv tiles 0 and 1 ----
#pragma unroll
    for (int i = 0; i < 16; ++i) {                // Q: 64 rows x 512B
        int idx = tid + i * NTHREADS;
        int r = idx >> 5, cc = (idx & 31) << 3;
        cp16(Qf + (b * DIM_N + q0 + r) * DIM_C + cc, sQ + r * QSTR + cc);
    }
    stage_kv(Kf, Vf, sK0, sV0, b, 0, tid);
    stage_kv(Kf, Vf, sK1, sV1, b, TK, tid);
    wait_async_16();          // Q + tile0 done (async retires in order); tile1 in flight
    __syncthreads();

    const int qrow = wave * 16 + lane15;
    v16h qf[8];
#pragma unroll
    for (int kc = 0; kc < 8; ++kc)
        qf[kc] = load_afrag(sQ + qrow * QSTR + kc * 32, hi8);

    v8f o[16];
#pragma unroll
    for (int cb = 0; cb < 16; ++cb) o[cb] = {};
    float m[8], l[8];
#pragma unroll
    for (int r = 0; r < 8; ++r) { m[r] = -1e30f; l[r] = 0.0f; }

    _Float16* sPw = sP + wave * 16 * PSTR;

    for (int it = 0; it < NT; ++it) {
        const _Float16* sKb = (it & 1) ? sK1 : sK0;
        const _Float16* sVb = (it & 1) ? sV1 : sV0;

        // ---- S = Q K^T (16x32 per wave), 16 WMMAs ----
        v8f s0 = {}, s1 = {};
#pragma unroll
        for (int kc = 0; kc < 8; ++kc) {
            v16h b0 = load_bfrag(sKb + lane15 * KSTR + kc * 32 + kb16);
            v16h b1 = load_bfrag(sKb + (16 + lane15) * KSTR + kc * 32 + kb16);
            s0 = wmma_f16(qf[kc], b0, s0);
            s1 = wmma_f16(qf[kc], b1, s1);
        }

        // ---- online softmax ----
        float p0[8], p1[8], alpha[8];
#pragma unroll
        for (int r = 0; r < 8; ++r) {
            float a0 = s0[r] * 0.0625f, a1 = s1[r] * 0.0625f;
            float mx = fmaxf(a0, a1);
            mx = fmaxf(mx, __shfl_xor(mx, 1));
            mx = fmaxf(mx, __shfl_xor(mx, 2));
            mx = fmaxf(mx, __shfl_xor(mx, 4));
            mx = fmaxf(mx, __shfl_xor(mx, 8));
            float mn = fmaxf(m[r], mx);
            alpha[r] = __expf(m[r] - mn);
            p0[r] = __expf(a0 - mn);
            p1[r] = __expf(a1 - mn);
            float rs = p0[r] + p1[r];
            rs += __shfl_xor(rs, 1);
            rs += __shfl_xor(rs, 2);
            rs += __shfl_xor(rs, 4);
            rs += __shfl_xor(rs, 8);
            l[r] = l[r] * alpha[r] + rs;
            m[r] = mn;
        }
#pragma unroll
        for (int cb = 0; cb < 16; ++cb)
#pragma unroll
            for (int r = 0; r < 8; ++r)
                o[cb][r] *= alpha[r];

        // ---- P: C/D layout -> A layout via wave-private LDS patch ----
#pragma unroll
        for (int r = 0; r < 8; ++r) {
            sPw[(r + hi8) * PSTR + lane15]      = (_Float16)p0[r];
            sPw[(r + hi8) * PSTR + 16 + lane15] = (_Float16)p1[r];
        }
        __builtin_amdgcn_wave_barrier();
        v16h pa = load_afrag(sPw + lane15 * PSTR, hi8);

        // ---- O += P V (16x256 per wave), 16 WMMAs ----
#pragma unroll
        for (int cb = 0; cb < 16; ++cb) {
            v16h bv = load_bfrag(sVb + (cb * 16 + lane15) * VSTR + kb16);
            o[cb] = wmma_f16(pa, bv, o[cb]);
        }

        // ---- pipeline: refill the buffer just consumed with tile it+2 ----
        __syncthreads();                       // all waves done reading buf[it&1]
        if (it + 2 < NT) {
            stage_kv(Kf, Vf, (it & 1) ? sK1 : sK0, (it & 1) ? sV1 : sV0,
                     b, (it + 2) * TK, tid);
            wait_async_16();                   // tile it+1 complete, it+2 in flight
        } else {
            wait_async_0();
        }
        if (it + 1 < NT) __syncthreads();
    }

    // ---- epilogue: normalize, transpose through LDS, coalesced store ----
    __syncthreads();
    float inv[8];
#pragma unroll
    for (int r = 0; r < 8; ++r) inv[r] = 1.0f / l[r];
#pragma unroll
    for (int cb = 0; cb < 16; ++cb)
#pragma unroll
        for (int r = 0; r < 8; ++r)
            sO[(wave * 16 + r + hi8) * OSTR + cb * 16 + lane15] = o[cb][r] * inv[r];
    __syncthreads();
    for (int idx = tid; idx < TQ * DIM_C; idx += NTHREADS) {
        int r = idx & (TQ - 1);
        int c = idx >> 6;
        gout[(b * DIM_C + c) * DIM_N + q0 + r] = sO[r * OSTR + c];
    }
}

// =======================================================================
// Fallback: fully fused kernel (round-1), used when workspace is too small
// =======================================================================
#define OFF_Q 0
#define OFF_K (TQ * QSTR * 2)
#define OFF_V (OFF_K + TK * KSTR * 2)
#define OFF_P (OFF_V + DIM_C * VSTR * 2)
#define SMEM_BYTES (OFF_P + NWAVE * 16 * PSTR * 2)

__global__ __launch_bounds__(NTHREADS)
void xattn_flash_fused(const float* __restrict__ gq, const float* __restrict__ gk,
                       const float* __restrict__ gv, const float* __restrict__ qpe,
                       const float* __restrict__ kpe, float* __restrict__ gout) {
    __shared__ __align__(16) char smem[SMEM_BYTES];
    _Float16* sQ = (_Float16*)(smem + OFF_Q);
    _Float16* sK = (_Float16*)(smem + OFF_K);
    _Float16* sV = (_Float16*)(smem + OFF_V);
    _Float16* sP = (_Float16*)(smem + OFF_P);
    float*    sO = (float*)smem;

    const int tid = threadIdx.x, wave = tid >> 5, lane = tid & 31;
    const int lane15 = lane & 15, hi8 = (lane >> 4) << 3, kb16 = (lane >> 4) << 4;
    const int b  = blockIdx.x / (DIM_N / TQ);
    const int q0 = (blockIdx.x % (DIM_N / TQ)) * TQ;

    for (int idx = tid; idx < TQ * DIM_C; idx += NTHREADS) {
        int r = idx & (TQ - 1), c = idx >> 6;
        float val = gq[(b * DIM_C + c) * DIM_N + q0 + r] + qpe[(q0 + r) * DIM_C + c];
        sQ[r * QSTR + c] = (_Float16)val;
    }
    __syncthreads();

    const int qrow = wave * 16 + lane15;
    v16h qf[8];
#pragma unroll
    for (int kc = 0; kc < 8; ++kc)
        qf[kc] = load_afrag(sQ + qrow * QSTR + kc * 32, hi8);

    v8f o[16];
#pragma unroll
    for (int cb = 0; cb < 16; ++cb) o[cb] = {};
    float m[8], l[8];
#pragma unroll
    for (int r = 0; r < 8; ++r) { m[r] = -1e30f; l[r] = 0.0f; }
    _Float16* sPw = sP + wave * 16 * PSTR;

    for (int k0 = 0; k0 < DIM_N; k0 += TK) {
        __syncthreads();
        for (int idx = tid; idx < TK * DIM_C; idx += NTHREADS) {
            int r = idx & (TK - 1), c = idx >> 5;
            int gidx = (b * DIM_C + c) * DIM_N + k0 + r;
            sK[r * KSTR + c] = (_Float16)(gk[gidx] + kpe[(k0 + r) * DIM_C + c]);
            sV[c * VSTR + r] = (_Float16)gv[gidx];
        }
        __syncthreads();

        v8f s0 = {}, s1 = {};
#pragma unroll
        for (int kc = 0; kc < 8; ++kc) {
            v16h b0 = load_bfrag(sK + lane15 * KSTR + kc * 32 + kb16);
            v16h b1 = load_bfrag(sK + (16 + lane15) * KSTR + kc * 32 + kb16);
            s0 = wmma_f16(qf[kc], b0, s0);
            s1 = wmma_f16(qf[kc], b1, s1);
        }
        float p0[8], p1[8], alpha[8];
#pragma unroll
        for (int r = 0; r < 8; ++r) {
            float a0 = s0[r] * 0.0625f, a1 = s1[r] * 0.0625f;
            float mx = fmaxf(a0, a1);
            mx = fmaxf(mx, __shfl_xor(mx, 1));
            mx = fmaxf(mx, __shfl_xor(mx, 2));
            mx = fmaxf(mx, __shfl_xor(mx, 4));
            mx = fmaxf(mx, __shfl_xor(mx, 8));
            float mn = fmaxf(m[r], mx);
            alpha[r] = __expf(m[r] - mn);
            p0[r] = __expf(a0 - mn);
            p1[r] = __expf(a1 - mn);
            float rs = p0[r] + p1[r];
            rs += __shfl_xor(rs, 1);
            rs += __shfl_xor(rs, 2);
            rs += __shfl_xor(rs, 4);
            rs += __shfl_xor(rs, 8);
            l[r] = l[r] * alpha[r] + rs;
            m[r] = mn;
        }
#pragma unroll
        for (int cb = 0; cb < 16; ++cb)
#pragma unroll
            for (int r = 0; r < 8; ++r)
                o[cb][r] *= alpha[r];
#pragma unroll
        for (int r = 0; r < 8; ++r) {
            sPw[(r + hi8) * PSTR + lane15]      = (_Float16)p0[r];
            sPw[(r + hi8) * PSTR + 16 + lane15] = (_Float16)p1[r];
        }
        __builtin_amdgcn_wave_barrier();
        v16h pa = load_afrag(sPw + lane15 * PSTR, hi8);
#pragma unroll
        for (int cb = 0; cb < 16; ++cb) {
            v16h bv = load_bfrag(sV + (cb * 16 + lane15) * VSTR + kb16);
            o[cb] = wmma_f16(pa, bv, o[cb]);
        }
    }
    __syncthreads();
    float inv[8];
#pragma unroll
    for (int r = 0; r < 8; ++r) inv[r] = 1.0f / l[r];
#pragma unroll
    for (int cb = 0; cb < 16; ++cb)
#pragma unroll
        for (int r = 0; r < 8; ++r)
            sO[(wave * 16 + r + hi8) * OSTR + cb * 16 + lane15] = o[cb][r] * inv[r];
    __syncthreads();
    for (int idx = tid; idx < TQ * DIM_C; idx += NTHREADS) {
        int r = idx & (TQ - 1), c = idx >> 6;
        gout[(b * DIM_C + c) * DIM_N + q0 + r] = sO[r * OSTR + c];
    }
}

// =======================================================================
extern "C" void kernel_launch(void* const* d_in, const int* in_sizes, int n_in,
                              void* d_out, int out_size, void* d_ws, size_t ws_size,
                              hipStream_t stream) {
    (void)in_sizes; (void)n_in; (void)out_size;
    const float* q   = (const float*)d_in[0];
    const float* k   = (const float*)d_in[1];
    const float* v   = (const float*)d_in[2];
    const float* qpe = (const float*)d_in[3];
    const float* kpe = (const float*)d_in[4];
    float* out = (float*)d_out;

    const size_t elems = (size_t)BATCH * DIM_N * DIM_C;        // 8388608
    const size_t need  = 3 * elems * sizeof(_Float16);          // 48 MiB

    if (ws_size >= need) {
        _Float16* Qf = (_Float16*)d_ws;
        _Float16* Kf = Qf + elems;
        _Float16* Vf = Kf + elems;
        dim3 tgrid(DIM_N / 64, DIM_C / 64, BATCH);              // 64 x 4 x 8
        hipLaunchKernelGGL(prep_qk, tgrid, dim3(256), 0, stream, q, qpe, Qf);
        hipLaunchKernelGGL(prep_qk, tgrid, dim3(256), 0, stream, k, kpe, Kf);
        hipLaunchKernelGGL(prep_v, dim3(elems / 1024), dim3(256), 0, stream, v, Vf);
        hipLaunchKernelGGL(xattn_flash_f16, dim3(BATCH * (DIM_N / TQ)), dim3(NTHREADS),
                           0, stream, Qf, Kf, Vf, out);
    } else {
        hipLaunchKernelGGL(xattn_flash_fused, dim3(BATCH * (DIM_N / TQ)), dim3(NTHREADS),
                           0, stream, q, k, v, qpe, kpe, out);
    }
}
